// MMDAlignment_20048907337956
// MI455X (gfx1250) — compile-verified, matrix operand order checked by reading, not run on hardware
//
#include <hip/hip_runtime.h>

typedef __attribute__((ext_vector_type(16))) __bf16 v16bf;
typedef __attribute__((ext_vector_type(8)))  __bf16 v8bf;
typedef __attribute__((ext_vector_type(4)))  __bf16 v4bf;
typedef __attribute__((ext_vector_type(8)))  float  v8f;

#define N_ROWS 8192
#define DIMS 128
#define TM_TILES 512            // 8192 / 16  (M tiles)
#define TN_GROUPS 128           // 8192 / 64  (N strips of 4 tiles)
#define GROUPS_PER_PAIR (TM_TILES * TN_GROUPS)
#define TOTAL_GROUPS (3 * GROUPS_PER_PAIR)
#define MAIN_BLOCKS 2048
#define WAVES_PER_BLOCK 8
#define PLANE_ELEMS ((size_t)N_ROWS * DIMS)   // 1M bf16 per plane

// ---- pre-pass: one wave per row; fused row-norm + fp32 -> (bf16 hi, lo) split
__global__ void mmd_prep(const float* __restrict__ sus,
                         const float* __restrict__ ccd,
                         float*  __restrict__ norms,   // 16384 floats
                         __bf16* __restrict__ hiS, __bf16* __restrict__ loS,
                         __bf16* __restrict__ hiC, __bf16* __restrict__ loC) {
  int wave = (blockIdx.x * blockDim.x + threadIdx.x) >> 5;
  int lane = threadIdx.x & 31;
  bool isS = wave < N_ROWS;
  const float* src = isS ? sus : ccd;
  __bf16* hip = isS ? hiS : hiC;
  __bf16* lop = isS ? loS : loC;
  int row = wave & (N_ROWS - 1);

  float4 v = ((const float4*)(src + (size_t)row * DIMS))[lane];  // 32 lanes * 4 = 128
  float fv[4] = {v.x, v.y, v.z, v.w};
  v4bf hi4, lo4;
  float s = 0.0f;
  #pragma unroll
  for (int j = 0; j < 4; ++j) {
    s += fv[j] * fv[j];
    __bf16 fh = (__bf16)fv[j];
    hi4[j] = fh;
    lo4[j] = (__bf16)(fv[j] - (float)fh);   // split-bf16 residual
  }
  size_t off = (size_t)row * DIMS + lane * 4;
  *(v4bf*)(hip + off) = hi4;
  *(v4bf*)(lop + off) = lo4;

  #pragma unroll
  for (int offd = 16; offd > 0; offd >>= 1) s += __shfl_down(s, offd, 32);
  if (lane == 0) norms[wave] = s;
}

// Assemble a 16x32 bf16 fragment from plain row-major [row][128] bf16 plane.
// ISA 16-bit A/B layout: lane half h selects K = kk+h*8+{0..7} (elems 0..7)
// and K = kk+16+h*8+{0..7} (elems 8..15). Both runs are contiguous 16B loads.
__device__ __forceinline__ v16bf load_bf_frag(const __bf16* __restrict__ rowp,
                                              int kk, int h) {
  v8bf a = *(const v8bf*)(rowp + kk + h * 8);
  v8bf b = *(const v8bf*)(rowp + kk + 16 + h * 8);
  return __builtin_shufflevector(a, b, 0, 1, 2, 3, 4, 5, 6, 7,
                                       8, 9, 10, 11, 12, 13, 14, 15);
}

// ---------------- main fused GEMM + RBF + reduce ----------------
__global__ void __launch_bounds__(256)
mmd_main(const __bf16* __restrict__ hiS, const __bf16* __restrict__ loS,
         const __bf16* __restrict__ hiC, const __bf16* __restrict__ loC,
         const float* __restrict__ norms,
         float* __restrict__ partials) {
  __shared__ float lds[WAVES_PER_BLOCK];
  const int lane = threadIdx.x & 31;
  const int h = lane >> 4;        // lane half (K sub-block select)
  const int r = lane & 15;        // row/col within tile
  const int wib = threadIdx.x >> 5;
  const int gwave = blockIdx.x * WAVES_PER_BLOCK + wib;
  const int stride = gridDim.x * WAVES_PER_BLOCK;

  float local = 0.0f;

  for (int g = gwave; g < TOTAL_GROUPS; g += stride) {
    const int p   = g / GROUPS_PER_PAIR;          // 0: SS, 1: CC, 2: SC
    const int rem = g - p * GROUPS_PER_PAIR;
    const int tm  = rem / TN_GROUPS;
    const int tn  = rem - tm * TN_GROUPS;

    const __bf16* Ahi = (p == 1) ? hiC : hiS;
    const __bf16* Alo = (p == 1) ? loC : loS;
    const __bf16* Bhi = (p == 0) ? hiS : hiC;
    const __bf16* Blo = (p == 0) ? loS : loC;
    const float* nA = norms + ((p == 1) ? N_ROWS : 0);
    const float* nB = norms + ((p == 0) ? 0 : N_ROWS);
    const float  w  = (p == 2) ? -2.0f : 1.0f;

    const int m0 = tm * 16;
    const int n0 = tn * 64;

    v8f acc[4] = {v8f{}, v8f{}, v8f{}, v8f{}};
    const size_t aoff = (size_t)(m0 + r) * DIMS;
    const size_t boff = (size_t)(n0 + r) * DIMS;
    const __bf16* arh = Ahi + aoff;
    const __bf16* arl = Alo + aoff;
    const __bf16* brh = Bhi + boff;
    const __bf16* brl = Blo + boff;

    #pragma unroll
    for (int kk = 0; kk < DIMS; kk += 32) {
      v16bf ahi = load_bf_frag(arh, kk, h);
      v16bf alo = load_bf_frag(arl, kk, h);
      #pragma unroll
      for (int t = 0; t < 4; ++t) {
        const size_t toff = (size_t)t * 16 * DIMS;
        v16bf bhi = load_bf_frag(brh + toff, kk, h);
        v16bf blo = load_bf_frag(brl + toff, kk, h);
        // split-bf16 fp32 emulation: hi*hi + hi*lo + lo*hi
        acc[t] = __builtin_amdgcn_wmma_f32_16x16x32_bf16(
            false, ahi, false, bhi, (short)0, acc[t], false, false);
        acc[t] = __builtin_amdgcn_wmma_f32_16x16x32_bf16(
            false, ahi, false, blo, (short)0, acc[t], false, false);
        acc[t] = __builtin_amdgcn_wmma_f32_16x16x32_bf16(
            false, alo, false, bhi, (short)0, acc[t], false, false);
      }
    }

    // C/D layout: VGPR rr, lane -> element (M = rr + 8*h, N = lane&15)
    float navs[8];
    #pragma unroll
    for (int rr = 0; rr < 8; ++rr) navs[rr] = nA[m0 + rr + 8 * h];

    float s = 0.0f;
    #pragma unroll
    for (int t = 0; t < 4; ++t) {
      const float nbv = nB[n0 + t * 16 + r];
      #pragma unroll
      for (int rr = 0; rr < 8; ++rr) {
        float d2 = navs[rr] + nbv - 2.0f * acc[t][rr];
        d2 = fmaxf(d2, 0.0f);
        s += __expf(-0.5f * d2);   // bandwidth = 1.0 -> /(2*bw^2) = 0.5
      }
    }
    local += w * s;
  }

  // deterministic block reduction (no float atomics)
  #pragma unroll
  for (int off = 16; off > 0; off >>= 1) local += __shfl_down(local, off, 32);
  if (lane == 0) lds[wib] = local;
  __syncthreads();
  if (threadIdx.x == 0) {
    float t = 0.0f;
    #pragma unroll
    for (int i = 0; i < WAVES_PER_BLOCK; ++i) t += lds[i];
    partials[blockIdx.x] = t;
  }
}

// ---------------- final deterministic reduction ----------------
__global__ void mmd_finalize(const float* __restrict__ partials, int n,
                             float* __restrict__ out) {
  __shared__ float lds[256];
  float s = 0.0f;
  for (int i = threadIdx.x; i < n; i += 256) s += partials[i];
  lds[threadIdx.x] = s;
  __syncthreads();
  for (int off = 128; off > 0; off >>= 1) {
    if ((int)threadIdx.x < off) lds[threadIdx.x] += lds[threadIdx.x + off];
    __syncthreads();
  }
  if (threadIdx.x == 0)
    out[0] = lds[0] * (1.0f / ((float)N_ROWS * (float)N_ROWS));
}

extern "C" void kernel_launch(void* const* d_in, const int* in_sizes, int n_in,
                              void* d_out, int out_size, void* d_ws, size_t ws_size,
                              hipStream_t stream) {
  const float* sus = (const float*)d_in[0];
  const float* ccd = (const float*)d_in[1];

  char* ws = (char*)d_ws;
  float*  norms    = (float*)ws;                                    // 64 KB
  __bf16* hiS      = (__bf16*)(ws + 65536);                         // 2 MB each
  __bf16* loS      = hiS + PLANE_ELEMS;
  __bf16* hiC      = loS + PLANE_ELEMS;
  __bf16* loC      = hiC + PLANE_ELEMS;
  float*  partials = (float*)(ws + 65536 + 4 * PLANE_ELEMS * sizeof(__bf16));

  mmd_prep<<<2048, 256, 0, stream>>>(sus, ccd, norms, hiS, loS, hiC, loC);
  mmd_main<<<MAIN_BLOCKS, 256, 0, stream>>>(hiS, loS, hiC, loC, norms, partials);
  mmd_finalize<<<1, 256, 0, stream>>>(partials, MAIN_BLOCKS, (float*)d_out);
}
